// CMoE_c_10591389352191
// MI455X (gfx1250) — matrix operand order repaired and checked by reading, not compiled, not run on hardware
//
#include <hip/hip_runtime.h>
#include <hip/hip_bf16.h>

// ---------------------------------------------------------------------------
// Hash-routed MoE (RWKV-style token-shift + relu^2 experts + sigmoid receptance)
// for MI455X / gfx1250: all GEMMs via v_wmma_f32_16x16x32_bf16 (wave32),
// double-buffered LDS staging exclusively via global_load_async_to_lds_b128
// (ASYNCcnt). Weights are downconverted AND transposed once per call so every
// GEMM sees its B operand as [N][K] row-major (contiguous in K).
// ---------------------------------------------------------------------------

typedef __attribute__((ext_vector_type(16))) __bf16 v16bf;
typedef __attribute__((ext_vector_type(8)))  float  v8f;

#define HASH_PRIME 5099

constexpr int Bdim = 8;
constexpr int Tdim = 2048;
constexpr int Ddim = 1024;
constexpr int Fdim = 2048;
constexpr int Edim = 16;
constexpr int Ndim = Bdim * Tdim;     // 16384 tokens
constexpr int Cdim = Ndim / Edim;     // 1024 capacity

// GEMM tiling
constexpr int BM = 128;
constexpr int BN = 128;
constexpr int BK = 32;
constexpr int LDA_SH = BK + 8;        // 40 bf16 -> 80B row stride (16B aligned)
constexpr int LDB_SH = BK + 8;        // B staged as [n][k]

struct bh4 { __bf16 a, b, c, d; };

// ---------------------------------------------------------------------------
// CDNA5 async global -> LDS helpers (tracked with ASYNCcnt).
// LDS destination VGPR carries the wave-relative LDS byte offset, which is
// the low 32 bits of the flat address of a __shared__ pointer.
// ---------------------------------------------------------------------------
__device__ __forceinline__ void async_copy_b128(void* lds_ptr, const void* gptr) {
  unsigned lds_off = (unsigned)(unsigned long long)lds_ptr;
  asm volatile("global_load_async_to_lds_b128 %0, %1, off"
               :: "v"(lds_off), "v"(gptr) : "memory");
}
__device__ __forceinline__ void wait_async0() {
  asm volatile("s_wait_asynccnt 0x0" ::: "memory");
}

// ---------------------------------------------------------------------------
// fp32 -> bf16 bulk conversion (vectorized, no transpose)
// ---------------------------------------------------------------------------
__global__ void f32_to_bf16_kernel(const float* __restrict__ src,
                                   __bf16* __restrict__ dst, int n4) {
  int i = blockIdx.x * blockDim.x + threadIdx.x;
  if (i < n4) {
    float4 v = ((const float4*)src)[i];
    bh4 h;
    h.a = (__bf16)v.x; h.b = (__bf16)v.y; h.c = (__bf16)v.z; h.d = (__bf16)v.w;
    ((bh4*)dst)[i] = h;
  }
}

// ---------------------------------------------------------------------------
// fp32 [R][C] -> bf16 [C][R] transpose + downconvert, 32x32 LDS tiles,
// per-z slab (one slab per expert). Block = 32x8.
// ---------------------------------------------------------------------------
__global__ void f32_to_bf16_tr_kernel(const float* __restrict__ src,
                                      __bf16* __restrict__ dst, int R, int C) {
  __shared__ float tile[32][33];
  const size_t slab = (size_t)R * C;
  src += (size_t)blockIdx.z * slab;
  dst += (size_t)blockIdx.z * slab;
  const int c0 = blockIdx.x * 32;
  const int r0 = blockIdx.y * 32;
  const int tx = threadIdx.x;          // 0..31
  const int ty = threadIdx.y;          // 0..7
#pragma unroll
  for (int i = 0; i < 32; i += 8)
    tile[ty + i][tx] = src[(size_t)(r0 + ty + i) * C + (c0 + tx)];
  __syncthreads();
#pragma unroll
  for (int i = 0; i < 32; i += 8)
    dst[(size_t)(c0 + ty + i) * R + (r0 + tx)] = (__bf16)tile[tx][ty + i];
}

// ---------------------------------------------------------------------------
// Deterministic hash routing. One block, 16 waves (wave32). Wave w owns
// expert w and scans all tokens in order, reproducing the reference cumsum
// with ballot/popc prefix sums.
// ---------------------------------------------------------------------------
__global__ void routing_kernel(const int* __restrict__ tok,
                               int* __restrict__ slot,
                               int* __restrict__ inv) {
  const int w    = threadIdx.x >> 5;   // expert id 0..15
  const int lane = threadIdx.x & 31;
  int cnt = 0;
  for (int base = 0; base < Ndim; base += 32) {
    int id = tok[base + lane];
    int e  = (id * HASH_PRIME) & (Edim - 1);   // ids < 50257 -> no overflow
    bool match = (e == w);
    unsigned mask = (unsigned)__ballot(match);
    int prefix = __popc(mask & ((1u << lane) - 1u));
    if (match) {
      int pos = cnt + prefix;
      int s = (pos < Cdim) ? (w * Cdim + pos) : (Edim * Cdim);
      slot[base + lane] = s;
      if (pos < Cdim) inv[w * Cdim + pos] = base + lane;
    }
    cnt += __popc(mask);
  }
  for (int p = cnt + lane; p < Cdim; p += 32) inv[w * Cdim + p] = -1;
}

// ---------------------------------------------------------------------------
// Token shift mix + dispatch. One block per token, 256 threads x float4.
// ---------------------------------------------------------------------------
__global__ void mix_dispatch_kernel(const float* __restrict__ x,
                                    const float* __restrict__ shift_state,
                                    const float* __restrict__ maa_k,
                                    const float* __restrict__ maa_r,
                                    const int* __restrict__ slot,
                                    __bf16* __restrict__ adisp,
                                    __bf16* __restrict__ xr) {
  const int n = blockIdx.x;
  const int b = n / Tdim;
  const int t = n - b * Tdim;
  const float4* xcur = (const float4*)(x + (size_t)n * Ddim);
  const float4* xprv = (t == 0) ? (const float4*)(shift_state + (size_t)b * Ddim)
                                : (const float4*)(x + (size_t)(n - 1) * Ddim);
  const int s = slot[n];
  const int d4 = threadIdx.x;                 // 256 * 4 = 1024 = D

  float4 xv = xcur[d4];
  float4 pv = xprv[d4];
  float4 mk = ((const float4*)maa_k)[d4];
  float4 mr = ((const float4*)maa_r)[d4];

  float dx0 = pv.x - xv.x, dx1 = pv.y - xv.y, dx2 = pv.z - xv.z, dx3 = pv.w - xv.w;

  bh4 hk, hr;
  hk.a = (__bf16)(xv.x + dx0 * mk.x); hk.b = (__bf16)(xv.y + dx1 * mk.y);
  hk.c = (__bf16)(xv.z + dx2 * mk.z); hk.d = (__bf16)(xv.w + dx3 * mk.w);
  hr.a = (__bf16)(xv.x + dx0 * mr.x); hr.b = (__bf16)(xv.y + dx1 * mr.y);
  hr.c = (__bf16)(xv.z + dx2 * mr.z); hr.d = (__bf16)(xv.w + dx3 * mr.w);

  ((bh4*)(xr + (size_t)n * Ddim))[d4] = hr;
  if (s < Edim * Cdim) ((bh4*)(adisp + (size_t)s * Ddim))[d4] = hk;
}

// ---------------------------------------------------------------------------
// Tiled bf16 WMMA GEMM, 256 threads (8 waves), 128x128 tile, K step 32,
// double-buffered LDS, staging entirely via global_load_async_to_lds_b128.
// A is [M][K] row-major; B is [N][K] row-major (pre-transposed weights).
// Wave grid 4x2; each wave -> 2 (M) x 4 (N) 16x16 fragments.
// MODE 0: D = relu(acc)^2 -> bf16 H             (per-expert z)
// MODE 1: scatter acc rows via inv[] -> float y  (per-expert z)
// MODE 2: out = sigmoid(acc) * yin               (dense)
// ---------------------------------------------------------------------------
template <int MODE>
__global__ void __launch_bounds__(256)
gemm_kernel(const __bf16* __restrict__ A, const __bf16* __restrict__ Bm,
            int K, int lda, int ldb,
            size_t aStrideZ, size_t bStrideZ,
            __bf16* __restrict__ Hout,
            float* __restrict__ Yout,
            const int* __restrict__ inv,
            const float* __restrict__ Yin) {
  __shared__ __bf16 Alds[2][BM * LDA_SH];
  __shared__ __bf16 Blds[2][BN * LDB_SH];     // [n][k]

  const int z   = blockIdx.z;
  const int n0  = blockIdx.x * BN;
  const int m0  = blockIdx.y * BM;
  const __bf16* Ag = A + (size_t)z * aStrideZ + (size_t)m0 * lda;
  const __bf16* Bg = Bm + (size_t)z * bStrideZ + (size_t)n0 * ldb;

  const int tid  = threadIdx.x;
  const int lane = tid & 31;
  const int wid  = tid >> 5;
  const int wm   = (wid >> 1) * 32;        // 0,32,64,96
  const int wn   = (wid & 1) * 64;         // 0,64

  // staging coordinates: thread -> one 16-elem row segment of each tile
  const int srow = tid >> 1;
  const int sseg = (tid & 1) * 16;

  auto stage = [&](int buf, int k0) {
    {
      const __bf16* src = Ag + (size_t)srow * lda + k0 + sseg;
      __bf16* dst = &Alds[buf][srow * LDA_SH + sseg];
      async_copy_b128(dst, src);
      async_copy_b128(dst + 8, src + 8);
    }
    {
      const __bf16* src = Bg + (size_t)srow * ldb + k0 + sseg;
      __bf16* dst = &Blds[buf][srow * LDB_SH + sseg];
      async_copy_b128(dst, src);
      async_copy_b128(dst + 8, src + 8);
    }
  };

  v8f acc[2][4];
#pragma unroll
  for (int i = 0; i < 2; ++i)
#pragma unroll
    for (int j = 0; j < 4; ++j)
#pragma unroll
      for (int q = 0; q < 8; ++q) acc[i][j][q] = 0.0f;

  const int hi    = (lane & 16) ? 1 : 0;
  const int koffA = hi ? 8 : 0;            // A: lanes 0-15 -> K{0..7,16..23}
  const int koffB = hi ? 8 : 0;            // B: lanes 0-15 -> K{0..7,16..23}
  const int frow  = lane & 15;

  // prologue: fill buffer 0
  stage(0, 0);
  wait_async0();
  __syncthreads();

  int cur = 0;
  for (int k0 = 0; k0 < K; k0 += BK) {
    const int knext = k0 + BK;
    if (knext < K) stage(cur ^ 1, knext);   // overlap staging with compute

    union Frag { v16bf v; uint4 q[2]; };
    v16bf afrag[2];
#pragma unroll
    for (int i = 0; i < 2; ++i) {
      const __bf16* ap = &Alds[cur][(wm + i * 16 + frow) * LDA_SH];
      Frag u;
      u.q[0] = *(const uint4*)(ap + koffA);        // K koffA..koffA+7
      u.q[1] = *(const uint4*)(ap + koffA + 16);   // K koffA+16..koffA+23
      afrag[i] = u.v;
    }
    v16bf bfrag[4];
#pragma unroll
    for (int j = 0; j < 4; ++j) {
      const __bf16* bp = &Blds[cur][(wn + j * 16 + frow) * LDB_SH + koffB];
      Frag u;
      u.q[0] = *(const uint4*)(bp);                // K koffB..koffB+7
      u.q[1] = *(const uint4*)(bp + 16);           // K koffB+16..koffB+23
      bfrag[j] = u.v;
    }
#pragma unroll
    for (int i = 0; i < 2; ++i)
#pragma unroll
      for (int j = 0; j < 4; ++j)
        acc[i][j] = __builtin_amdgcn_wmma_f32_16x16x32_bf16(
            false, afrag[i], false, bfrag[j], (short)0, acc[i][j], false, false);

    wait_async0();        // own async stores to LDS visible before barrier
    __syncthreads();
    cur ^= 1;
  }

  // --- epilogue ---
  const int mbase = m0 + wm + (hi ? 8 : 0);  // C/D frag: M = elem + (lane>=16 ? 8:0)
  const int nbase = n0 + wn + frow;          // C/D frag: N = lane & 15

  if (MODE == 0) {
    __bf16* Hz = Hout + (size_t)z * Cdim * Fdim;
#pragma unroll
    for (int i = 0; i < 2; ++i)
#pragma unroll
      for (int e = 0; e < 8; ++e) {
        int r = mbase + i * 16 + e;
#pragma unroll
        for (int j = 0; j < 4; ++j) {
          float v = acc[i][j][e];
          v = (v > 0.0f) ? v * v : 0.0f;
          Hz[(size_t)r * Fdim + nbase + j * 16] = (__bf16)v;
        }
      }
  } else if (MODE == 1) {
    const int* invz = inv + z * Cdim;        // row indices are expert-local
#pragma unroll
    for (int i = 0; i < 2; ++i)
#pragma unroll
      for (int e = 0; e < 8; ++e) {
        int r = mbase + i * 16 + e;          // local row within expert (< C)
        int n = invz[r];
        if (n >= 0) {
#pragma unroll
          for (int j = 0; j < 4; ++j)
            Yout[(size_t)n * Ddim + nbase + j * 16] = acc[i][j][e];
        }
      }
  } else {
#pragma unroll
    for (int i = 0; i < 2; ++i)
#pragma unroll
      for (int e = 0; e < 8; ++e) {
        int r = mbase + i * 16 + e;          // token index
#pragma unroll
        for (int j = 0; j < 4; ++j) {
          float g = 1.0f / (1.0f + __expf(-acc[i][j][e]));
          size_t idx = (size_t)r * Ddim + nbase + j * 16;
          Yout[idx] = g * Yin[idx];
        }
      }
  }
}

// ---------------------------------------------------------------------------
// Launcher
// ---------------------------------------------------------------------------
extern "C" void kernel_launch(void* const* d_in, const int* in_sizes, int n_in,
                              void* d_out, int out_size, void* d_ws, size_t ws_size,
                              hipStream_t stream) {
  const float* x     = (const float*)d_in[0];
  const int*   tok   = (const int*)  d_in[1];
  const float* shift = (const float*)d_in[2];
  const float* maa_k = (const float*)d_in[3];
  const float* maa_r = (const float*)d_in[4];
  const float* w_rec = (const float*)d_in[5];
  const float* w_key = (const float*)d_in[6];
  const float* w_val = (const float*)d_in[7];
  float* out = (float*)d_out;

  char* ws = (char*)d_ws;
  size_t off = 0;
  int*    slot  = (int*)(ws + off);    off += (size_t)Ndim * 4;
  int*    inv   = (int*)(ws + off);    off += (size_t)Edim * Cdim * 4;
  __bf16* adisp = (__bf16*)(ws + off); off += (size_t)Edim * Cdim * Ddim * 2;
  __bf16* xr    = (__bf16*)(ws + off); off += (size_t)Ndim * Ddim * 2;
  __bf16* Hbuf  = (__bf16*)(ws + off); off += (size_t)Edim * Cdim * Fdim * 2;
  float*  ybuf  = (float*)(ws + off);  off += (size_t)Ndim * Ddim * 4;
  __bf16* wkT   = (__bf16*)(ws + off); off += (size_t)Edim * Ddim * Fdim * 2; // [E][F][D]
  __bf16* wvT   = (__bf16*)(ws + off); off += (size_t)Edim * Fdim * Ddim * 2; // [E][D][F]
  __bf16* wr    = (__bf16*)(ws + off); off += (size_t)Ddim * Ddim * 2;        // [D][D] ([N][K])
  (void)off; (void)ws_size; (void)in_sizes; (void)n_in; (void)out_size;

  // 1) weight fp32 -> bf16, transposed to [N][K] per expert
  f32_to_bf16_tr_kernel<<<dim3(Fdim / 32, Ddim / 32, Edim), dim3(32, 8), 0, stream>>>(
      w_key, wkT, Ddim, Fdim);             // [D][F] -> [F][D]
  f32_to_bf16_tr_kernel<<<dim3(Ddim / 32, Fdim / 32, Edim), dim3(32, 8), 0, stream>>>(
      w_val, wvT, Fdim, Ddim);             // [F][D] -> [D][F]
  {
    int m4 = Ddim * Ddim / 4;              // w_recept rows are already N: no transpose
    f32_to_bf16_kernel<<<(m4 + 255) / 256, 256, 0, stream>>>(w_rec, wr, m4);
  }

  // 2) deterministic hash routing
  routing_kernel<<<1, 512, 0, stream>>>(tok, slot, inv);

  // 3) token-shift mix + dispatch
  mix_dispatch_kernel<<<Ndim, 256, 0, stream>>>(x, shift, maa_k, maa_r, slot, adisp, xr);

  // 4) zero combine buffer (dropped tokens contribute 0)
  hipMemsetAsync(ybuf, 0, (size_t)Ndim * Ddim * 4, stream);

  // 5) GEMM1: H = relu(In @ W1)^2   [per expert: (C x D) @ (D x F)]
  gemm_kernel<0><<<dim3(Fdim / BN, Cdim / BM, Edim), 256, 0, stream>>>(
      adisp, wkT, Ddim, Ddim, Ddim,
      (size_t)Cdim * Ddim, (size_t)Ddim * Fdim,
      Hbuf, nullptr, nullptr, nullptr);

  // 6) GEMM2: y[inv[row]] = H @ W2  [per expert: (C x F) @ (F x D)]
  gemm_kernel<1><<<dim3(Ddim / BN, Cdim / BM, Edim), 256, 0, stream>>>(
      Hbuf, wvT, Fdim, Fdim, Fdim,
      (size_t)Cdim * Fdim, (size_t)Fdim * Ddim,
      nullptr, ybuf, inv, nullptr);

  // 7) GEMM3: out = sigmoid(xr @ Wr^T) * y   [(N x D) @ (D x D)]
  gemm_kernel<2><<<dim3(Ddim / BN, Ndim / BM, 1), 256, 0, stream>>>(
      xr, wr, Ddim, Ddim, Ddim,
      0, 0,
      nullptr, out, nullptr, ybuf);
}
